// FvdbBasicBlock_8804682957040
// MI455X (gfx1250) — matrix operand order repaired and checked by reading, not compiled, not run on hardware
//
#include <hip/hip_runtime.h>

// ---------------------------------------------------------------------------
// FvdbBasicBlock on MI455X (gfx1250, wave32, WMMA).
// ~142 GFLOP of 128x128-K GEMMs (54 gathered sparse-conv taps dominate);
// working set fits the 192MB L2, so the kernel is matrix-pipe/LDS bound, not
// HBM bound -> bf16 WMMA (v_wmma_f32_16x16x32_bf16, f32 accum).
// Round 3: 4-wave blocks with 32 output rows per wave (2 row-groups sharing
// every B fragment -> 1.25 ds_load_b128/WMMA instead of 2.25) and a manual
// 1-deep software pipeline on B fragments so dscnt waits are partial and
// amortized over 2 WMMAs. sconv keeps double-buffered LDS tiles.
// ---------------------------------------------------------------------------

#define NVOX  65536
#define CCH   128
#define NSEGC 1024
#define NTAP  27
#define LDSP  144   // LDS row pitch in bf16 elems (288B, multiple of 32B)

typedef __bf16 bf16_t;
typedef __attribute__((ext_vector_type(16))) __bf16 v16bf;
typedef __attribute__((ext_vector_type(8)))  __bf16 v8bf;
typedef __attribute__((ext_vector_type(8)))  float  v8f;

// ------------------------- float max <-> orderable uint ---------------------
__device__ __forceinline__ unsigned f2ord(float f) {
  unsigned u = __float_as_uint(f);
  return (u & 0x80000000u) ? ~u : (u | 0x80000000u);
}
__device__ __forceinline__ float ord2f(unsigned u) {
  return (u & 0x80000000u) ? __uint_as_float(u & 0x7fffffffu)
                           : __uint_as_float(~u);
}

// ------------------------- WMMA tile helpers --------------------------------
// A frag (16x32 bf16): lanes 0-15 -> K {k0..k0+15},
// lanes 16-31 -> K {k0+16..k0+31}. B frag (32x16): lane = N col,
// half selects K 0-15 vs 16-31 (contiguous 32B in the K-major LDS row).
__device__ __forceinline__ v16bf ldsA_frag(const bf16_t* arow, int k0, int hf) {
  v8bf lo = *(const v8bf*)(arow + k0 + 16 * hf);
  v8bf hi = *(const v8bf*)(arow + k0 + 16 * hf + 8);
  return __builtin_shufflevector(lo, hi, 0, 1, 2, 3, 4, 5, 6, 7, 8, 9, 10, 11,
                                 12, 13, 14, 15);
}

// Each wave owns 32 output rows (two 16-row groups) x 128 cols.  Every B
// fragment is used by two WMMAs, and the next B fragment is loaded before the
// current pair issues (1-deep pipeline) so its LDS latency hides under them.
__device__ __forceinline__ void mma_tile(const bf16_t* sA, const bf16_t* sB,
                                         v8f acc0[8], v8f acc1[8], int wave,
                                         int hf, int r) {
  const bf16_t* arow0 = sA + (wave * 32 + r) * LDSP;
  const bf16_t* arow1 = arow0 + 16 * LDSP;
  const bf16_t* bbase = sB + r * LDSP + 16 * hf;
#pragma unroll
  for (int kc = 0; kc < 4; ++kc) {
    const int k0 = kc * 32;
    v16bf a0 = ldsA_frag(arow0, k0, hf);
    v16bf a1 = ldsA_frag(arow1, k0, hf);
    v16bf bcur = *(const v16bf*)(bbase + k0);
#pragma unroll
    for (int ct = 0; ct < 8; ++ct) {
      v16bf bnext = bcur;
      if (ct < 7) bnext = *(const v16bf*)(bbase + (ct + 1) * 16 * LDSP + k0);
      acc0[ct] = __builtin_amdgcn_wmma_f32_16x16x32_bf16(
          false, a0, false, bcur, (short)0, acc0[ct], false, false);
      acc1[ct] = __builtin_amdgcn_wmma_f32_16x16x32_bf16(
          false, a1, false, bcur, (short)0, acc1[ct], false, false);
      bcur = bnext;
    }
  }
}

__device__ __forceinline__ void store_rows(float* __restrict__ Y, int baserow,
                                           int hf, int r, const v8f acc[8]) {
  const int grow = baserow + 8 * hf;  // D layout: VGPR v -> M = v + 8*hf
#pragma unroll
  for (int ct = 0; ct < 8; ++ct) {
    const int col = ct * 16 + r;
#pragma unroll
    for (int v = 0; v < 8; ++v)
      Y[(size_t)(grow + v) * CCH + col] = acc[ct][v];
  }
}

// sB[c][k] <- Wt[c*kstride + kb + k]  (Wt pre-transposed bf16; 128 threads)
__device__ __forceinline__ void load_btile(const bf16_t* __restrict__ Wt,
                                           bf16_t* sB, int tid, int kstride,
                                           int kb) {
  for (int e = tid; e < CCH * 16; e += 128) {
    const int cc = e >> 4;
    const int kk = (e & 15) * 8;
    *(v8bf*)(sB + cc * LDSP + kk) =
        *(const v8bf*)(Wt + (size_t)cc * kstride + kb + kk);
  }
}

// ------------------------- dense GEMM: Y[N,128] = X[N,Ktot] @ W ------------
__global__ __launch_bounds__(128) void gemm_bf16_k(
    const float* __restrict__ X, const bf16_t* __restrict__ Wt,
    float* __restrict__ Y, int Ktot) {
  __shared__ __attribute__((aligned(32))) bf16_t sA[128 * LDSP];
  __shared__ __attribute__((aligned(32))) bf16_t sB[128 * LDSP];
  const int tid = threadIdx.x;
  const int wave = tid >> 5, lane = tid & 31;
  const int hf = lane >> 4, r = lane & 15;
  const int row0 = blockIdx.x * 128;
  v8f acc0[8], acc1[8];
#pragma unroll
  for (int t = 0; t < 8; ++t)
#pragma unroll
    for (int v = 0; v < 8; ++v) { acc0[t][v] = 0.f; acc1[t][v] = 0.f; }

  for (int kb = 0; kb < Ktot; kb += 128) {
    // A tile fp32 -> bf16: one thread per row, coalesced float4 reads
    {
      const float* src = X + (size_t)(row0 + tid) * Ktot + kb;
      bf16_t* dst = sA + tid * LDSP;
#pragma unroll
      for (int c = 0; c < 128; c += 4) {
        float4 vv = *(const float4*)(src + c);
        dst[c + 0] = (bf16_t)vv.x; dst[c + 1] = (bf16_t)vv.y;
        dst[c + 2] = (bf16_t)vv.z; dst[c + 3] = (bf16_t)vv.w;
      }
    }
    load_btile(Wt, sB, tid, Ktot, kb);
    __syncthreads();
    mma_tile(sA, sB, acc0, acc1, wave, hf, r);
    __syncthreads();
  }
  store_rows(Y, row0 + wave * 32, hf, r, acc0);
  store_rows(Y, row0 + wave * 32 + 16, hf, r, acc1);
}

// --------- sparse conv: Y = sum_k gather(X, idx[:,k])*mask @ W[k] -----------
// Double-buffered LDS (2 x (36KB A + 36KB B) = 144KB/WG, 2 WGs/WGP): tap t+1's
// gather + weight tile stream into the idle buffer while tap t's WMMAs run.
__global__ __launch_bounds__(128) void sconv_bf16_k(
    const float* __restrict__ X, const int* __restrict__ nidx,
    const bf16_t* __restrict__ WtAll, float* __restrict__ Y) {
  __shared__ __attribute__((aligned(32))) bf16_t sA[2][128 * LDSP];
  __shared__ __attribute__((aligned(32))) bf16_t sB[2][128 * LDSP];
  const int tid = threadIdx.x;
  const int wave = tid >> 5, lane = tid & 31;
  const int hf = lane >> 4, r = lane & 15;
  const int row0 = blockIdx.x * 128;
  v8f acc0[8], acc1[8];
#pragma unroll
  for (int t = 0; t < 8; ++t)
#pragma unroll
    for (int v = 0; v < 8; ++v) { acc0[t][v] = 0.f; acc1[t][v] = 0.f; }

  auto loadA = [&](bf16_t* base, int tap) {  // one thread per gathered row
    const int g = nidx[(size_t)(row0 + tid) * NTAP + tap];
    const bool ok = ((unsigned)g < (unsigned)NVOX);
    const float* src = X + (size_t)(ok ? g : 0) * CCH;
    bf16_t* dst = base + tid * LDSP;
#pragma unroll
    for (int c = 0; c < 128; c += 4) {
      float4 vv = *(const float4*)(src + c);
      if (!ok) { vv.x = 0.f; vv.y = 0.f; vv.z = 0.f; vv.w = 0.f; }
      dst[c + 0] = (bf16_t)vv.x; dst[c + 1] = (bf16_t)vv.y;
      dst[c + 2] = (bf16_t)vv.z; dst[c + 3] = (bf16_t)vv.w;
    }
  };

  loadA(sA[0], 0);
  load_btile(WtAll, sB[0], tid, CCH, 0);
  __syncthreads();
  for (int tap = 0; tap < NTAP; ++tap) {
    const int cur = tap & 1;
    if (tap + 1 < NTAP) {
      loadA(sA[cur ^ 1], tap + 1);
      load_btile(WtAll + (size_t)(tap + 1) * CCH * CCH, sB[cur ^ 1], tid, CCH, 0);
    }
    mma_tile(sA[cur], sB[cur], acc0, acc1, wave, hf, r);
    __syncthreads();
  }
  store_rows(Y, row0 + wave * 32, hf, r, acc0);
  store_rows(Y, row0 + wave * 32 + 16, hf, r, acc1);
}

// ------------------ weight convert+transpose: dst[m][c][k] = src[m][k][c] ---
__global__ void wconvT_k(const float* __restrict__ src, bf16_t* __restrict__ dst,
                         int nmat, int K, int Cc) {
  const int i = blockIdx.x * blockDim.x + threadIdx.x;
  const int per = K * Cc;
  if (i >= nmat * per) return;
  const int m = i / per;
  const int rk = i % per;
  const int k = rk / Cc;
  const int c = rk % Cc;
  dst[(size_t)m * per + (size_t)c * K + k] = (bf16_t)src[i];
}

// ------------------ column BN stats -> scale/shift (biased var) -------------
__global__ __launch_bounds__(256) void colstats_k(const float* __restrict__ X,
                                                  const float* __restrict__ g,
                                                  const float* __restrict__ b,
                                                  float* __restrict__ ss) {
  const int c = blockIdx.x;
  const int tid = threadIdx.x;
  float s = 0.f, q = 0.f;
  for (int n = tid; n < NVOX; n += 256) {
    const float v = X[(size_t)n * CCH + c];
    s += v; q += v * v;
  }
  __shared__ float sm[256], qm[256];
  sm[tid] = s; qm[tid] = q;
  __syncthreads();
  for (int o = 128; o > 0; o >>= 1) {
    if (tid < o) { sm[tid] += sm[tid + o]; qm[tid] += qm[tid + o]; }
    __syncthreads();
  }
  if (tid == 0) {
    const float m = sm[0] * (1.f / NVOX);
    const float var = qm[0] * (1.f / NVOX) - m * m;
    const float sc = g[c] * rsqrtf(var + 1e-5f);
    ss[c] = sc;
    ss[CCH + c] = b[c] - m * sc;
  }
}

// y = lrelu(x*scale+shift [+pre]) [+post]
__global__ void bn_act_k(const float* __restrict__ X, const float* __restrict__ ss,
                         const float* __restrict__ pre,
                         const float* __restrict__ post, float* __restrict__ Y,
                         int total) {
  const int i = blockIdx.x * blockDim.x + threadIdx.x;
  if (i >= total) return;
  const int c = i & (CCH - 1);
  float v = X[i] * ss[c] + ss[CCH + c];
  if (pre) v += pre[i];
  v = (v >= 0.f) ? v : 0.01f * v;
  if (post) v += post[i];
  Y[i] = v;
}

__global__ void fill_f_k(float* p, float v, int n) {
  const int i = blockIdx.x * blockDim.x + threadIdx.x;
  if (i < n) p[i] = v;
}
__global__ void fill_u_k(unsigned* p, unsigned v, int n) {
  const int i = blockIdx.x * blockDim.x + threadIdx.x;
  if (i < n) p[i] = v;
}
__global__ void seg_count_k(const int* __restrict__ cl, float* __restrict__ cnt) {
  const int i = blockIdx.x * blockDim.x + threadIdx.x;
  if (i < NVOX) atomicAdd(&cnt[cl[i]], 1.f);
}
__global__ void seg_sum_k(const float* __restrict__ X, const int* __restrict__ cl,
                          float* __restrict__ out) {
  const int i = blockIdx.x * blockDim.x + threadIdx.x;
  if (i >= NVOX * CCH) return;
  atomicAdd(&out[(size_t)cl[i >> 7] * CCH + (i & (CCH - 1))], X[i]);
}
__global__ void seg_center_k(float* __restrict__ X, const int* __restrict__ cl,
                             const float* __restrict__ segsum,
                             const float* __restrict__ cnt) {
  const int i = blockIdx.x * blockDim.x + threadIdx.x;
  if (i >= NVOX * CCH) return;
  const int s = cl[i >> 7];
  X[i] -= segsum[(size_t)s * CCH + (i & (CCH - 1))] / fmaxf(cnt[s], 1.f);
}
__global__ void seg_norm_k(float* __restrict__ X, const int* __restrict__ cl,
                           const float* __restrict__ segsum) {
  const int i = blockIdx.x * blockDim.x + threadIdx.x;
  if (i >= NVOX * CCH) return;
  X[i] /= (segsum[(size_t)cl[i >> 7] * CCH + (i & (CCH - 1))] + 1e-6f);
}
__global__ __launch_bounds__(256) void reduce_max_k(const float* __restrict__ X,
                                                    int total,
                                                    unsigned* __restrict__ gmax) {
  const int i = blockIdx.x * blockDim.x + threadIdx.x;
  float m = -3.0e38f;
  for (int j = i; j < total; j += gridDim.x * blockDim.x) m = fmaxf(m, X[j]);
  __shared__ float sm[256];
  sm[threadIdx.x] = m;
  __syncthreads();
  for (int o = 128; o > 0; o >>= 1) {
    if (threadIdx.x < o) sm[threadIdx.x] = fmaxf(sm[threadIdx.x], sm[threadIdx.x + o]);
    __syncthreads();
  }
  if (threadIdx.x == 0) atomicMax(gmax, f2ord(sm[0]));
}
__global__ void exp_sub_k(float* __restrict__ X, const unsigned* __restrict__ gmax,
                          int total) {
  const int i = blockIdx.x * blockDim.x + threadIdx.x;
  if (i >= total) return;
  X[i] = expf(X[i] - ord2f(*gmax));
}
__global__ void mul_ip_k(float* __restrict__ X, const float* __restrict__ Y,
                         int total) {
  const int i = blockIdx.x * blockDim.x + threadIdx.x;
  if (i < total) X[i] *= Y[i];
}
__global__ __launch_bounds__(256) void adaptive_k(const float* __restrict__ feat,
                                                  const float* __restrict__ aw,
                                                  float* __restrict__ adp) {
  __shared__ float w[CCH * 3];
  for (int i = threadIdx.x; i < CCH * 3; i += blockDim.x) w[i] = aw[i];
  __syncthreads();
  const int n = blockIdx.x * blockDim.x + threadIdx.x;
  if (n >= NVOX) return;
  float t0 = 0.f, t1 = 0.f, t2 = 0.f;
  for (int c = 0; c < CCH; ++c) {
    const float f = feat[(size_t)n * CCH + c];
    t0 += f * w[c * 3 + 0]; t1 += f * w[c * 3 + 1]; t2 += f * w[c * 3 + 2];
  }
  const float m = fmaxf(t0, fmaxf(t1, t2));
  const float e0 = expf(t0 - m), e1 = expf(t1 - m), e2 = expf(t2 - m);
  const float inv = 1.f / (e0 + e1 + e2);
  adp[(size_t)n * 3 + 0] = e0 * inv;
  adp[(size_t)n * 3 + 1] = e1 * inv;
  adp[(size_t)n * 3 + 2] = e2 * inv;
}
__global__ void concat_k(const float* __restrict__ flast,
                         const float* __restrict__ adp,
                         const float* __restrict__ fseg,
                         const int* __restrict__ clusters,
                         float* __restrict__ out) {
  const int i = blockIdx.x * blockDim.x + threadIdx.x;
  if (i >= NVOX * CCH) return;
  const int n = i >> 7, c = i & (CCH - 1);
  out[(size_t)n * 256 + c] = flast[i];
  float a = 0.f;
#pragma unroll
  for (int l = 0; l < 3; ++l) {
    const int cl = clusters[(size_t)l * NVOX + n];
    a += adp[(size_t)n * 3 + l] * fseg[((size_t)l * NSEGC + cl) * CCH + c];
  }
  out[(size_t)n * 256 + CCH + c] = a;
}

// ---------------------------------------------------------------------------
extern "C" void kernel_launch(void* const* d_in, const int* in_sizes, int n_in,
                              void* d_out, int out_size, void* d_ws,
                              size_t ws_size, hipStream_t stream) {
  (void)in_sizes; (void)n_in; (void)out_size; (void)ws_size;
  const float* feat       = (const float*)d_in[0];
  const int*   clusters   = (const int*)d_in[1];
  const int*   nidx       = (const int*)d_in[2];
  const float* lw_w       = (const float*)d_in[3];
  const float* lw_gamma   = (const float*)d_in[4];
  const float* lw_beta    = (const float*)d_in[5];
  const float* w_w        = (const float*)d_in[6];
  const float* proj_w     = (const float*)d_in[7];
  const float* proj_gamma = (const float*)d_in[8];
  const float* proj_beta  = (const float*)d_in[9];
  const float* adaptive_w = (const float*)d_in[10];
  const float* fuse_w     = (const float*)d_in[11];
  const float* fuse_gamma = (const float*)d_in[12];
  const float* fuse_beta  = (const float*)d_in[13];
  const float* conv1_w    = (const float*)d_in[14];
  const float* conv2_w    = (const float*)d_in[15];
  const float* bn1_gamma  = (const float*)d_in[16];
  const float* bn1_beta   = (const float*)d_in[17];
  const float* bn2_gamma  = (const float*)d_in[18];
  const float* bn2_beta   = (const float*)d_in[19];

  char* ws = (char*)d_ws;
  size_t off = 0;
  auto alloc = [&](size_t b) -> void* {
    void* p = ws + off;
    off = (off + b + 255) & ~(size_t)255;
    return p;
  };
  const size_t MAT = (size_t)CCH * CCH;  // 16384
  bf16_t* wt_lw   = (bf16_t*)alloc(3 * MAT * 2);
  bf16_t* wt_ww   = (bf16_t*)alloc(3 * MAT * 2);
  bf16_t* wt_proj = (bf16_t*)alloc(4 * MAT * 2);
  bf16_t* wt_fuse = (bf16_t*)alloc(2 * MAT * 2);
  bf16_t* wt_c1   = (bf16_t*)alloc((size_t)NTAP * MAT * 2);
  bf16_t* wt_c2   = (bf16_t*)alloc((size_t)NTAP * MAT * 2);
  const size_t NCB = (size_t)NVOX * CCH * 4;  // 33.5 MB
  float* bufA   = (float*)alloc(NCB);
  float* bufB   = (float*)alloc(NCB);
  float* fused  = (float*)alloc(NCB);
  float* d1     = (float*)alloc(NCB);
  float* ctmp   = (float*)alloc(NCB);
  float* concat = (float*)alloc((size_t)NVOX * 256 * 4);
  float* fseg   = (float*)alloc((size_t)3 * NSEGC * CCH * 4);
  float* segtmp = (float*)alloc((size_t)NSEGC * CCH * 4);
  float* cnt    = (float*)alloc((size_t)NSEGC * 4);
  float* adp    = (float*)alloc((size_t)NVOX * 3 * 4);
  float* stats  = (float*)alloc(2 * CCH * 4);
  unsigned* gmax = (unsigned*)alloc(64);

  const int NC = NVOX * CCH;
  const dim3 T(256);
  const dim3 TG(128);
  auto G = [](int n) { return dim3((unsigned)((n + 255) / 256)); };
  const dim3 GEMM_G(NVOX / 128);
  const dim3 COL_G(CCH);

  // ---- weight transpose + bf16 convert -------------------------------------
  wconvT_k<<<G(3 * (int)MAT), T, 0, stream>>>(lw_w, wt_lw, 3, 128, 128);
  wconvT_k<<<G(3 * (int)MAT), T, 0, stream>>>(w_w, wt_ww, 3, 128, 128);
  wconvT_k<<<G(4 * (int)MAT), T, 0, stream>>>(proj_w, wt_proj, 4, 128, 128);
  wconvT_k<<<G(2 * (int)MAT), T, 0, stream>>>(fuse_w, wt_fuse, 1, 256, 128);
  wconvT_k<<<G(NTAP * (int)MAT), T, 0, stream>>>(conv1_w, wt_c1, NTAP, 128, 128);
  wconvT_k<<<G(NTAP * (int)MAT), T, 0, stream>>>(conv2_w, wt_c2, NTAP, 128, 128);

  // ---- hierarchical levels --------------------------------------------------
  for (int i = 0; i < 3; ++i) {
    const int* cl = clusters + (size_t)i * NVOX;
    // pw = lrelu(bn(feat @ lw_w[i]))
    gemm_bf16_k<<<GEMM_G, TG, 0, stream>>>(feat, wt_lw + i * MAT, bufA, 128);
    colstats_k<<<COL_G, T, 0, stream>>>(bufA, lw_gamma + i * CCH, lw_beta + i * CCH, stats);
    bn_act_k<<<G(NC), T, 0, stream>>>(bufA, stats, nullptr, nullptr, bufA, NC);
    // per-cluster centering
    fill_f_k<<<G(NSEGC * CCH), T, 0, stream>>>(segtmp, 0.f, NSEGC * CCH);
    fill_f_k<<<G(NSEGC), T, 0, stream>>>(cnt, 0.f, NSEGC);
    seg_count_k<<<G(NVOX), T, 0, stream>>>(cl, cnt);
    seg_sum_k<<<G(NC), T, 0, stream>>>(bufA, cl, segtmp);
    seg_center_k<<<G(NC), T, 0, stream>>>(bufA, cl, segtmp, cnt);
    // pw = exp(pw @ w_w - globalmax) / (cluster segsum + 1e-6)
    gemm_bf16_k<<<GEMM_G, TG, 0, stream>>>(bufA, wt_ww + i * MAT, bufB, 128);
    fill_u_k<<<dim3(1), T, 0, stream>>>(gmax, 0u, 1);
    reduce_max_k<<<dim3(512), T, 0, stream>>>(bufB, NC, gmax);
    exp_sub_k<<<G(NC), T, 0, stream>>>(bufB, gmax, NC);
    fill_f_k<<<G(NSEGC * CCH), T, 0, stream>>>(segtmp, 0.f, NSEGC * CCH);
    seg_sum_k<<<G(NC), T, 0, stream>>>(bufB, cl, segtmp);
    seg_norm_k<<<G(NC), T, 0, stream>>>(bufB, cl, segtmp);
    // pf = lrelu(bn(feat @ proj_w[i])) * pw ; feats_seg[i] = segsum(pf)
    gemm_bf16_k<<<GEMM_G, TG, 0, stream>>>(feat, wt_proj + i * MAT, bufA, 128);
    colstats_k<<<COL_G, T, 0, stream>>>(bufA, proj_gamma + i * CCH, proj_beta + i * CCH, stats);
    bn_act_k<<<G(NC), T, 0, stream>>>(bufA, stats, nullptr, nullptr, bufA, NC);
    mul_ip_k<<<G(NC), T, 0, stream>>>(bufA, bufB, NC);
    fill_f_k<<<G(NSEGC * CCH), T, 0, stream>>>(fseg + (size_t)i * NSEGC * CCH, 0.f, NSEGC * CCH);
    seg_sum_k<<<G(NC), T, 0, stream>>>(bufA, cl, fseg + (size_t)i * NSEGC * CCH);
  }

  // ---- adaptive weights + fuse ---------------------------------------------
  adaptive_k<<<G(NVOX), T, 0, stream>>>(feat, adaptive_w, adp);
  gemm_bf16_k<<<GEMM_G, TG, 0, stream>>>(feat, wt_proj + 3 * MAT, bufA, 128);
  colstats_k<<<COL_G, T, 0, stream>>>(bufA, proj_gamma + 3 * CCH, proj_beta + 3 * CCH, stats);
  bn_act_k<<<G(NC), T, 0, stream>>>(bufA, stats, nullptr, nullptr, bufA, NC);
  concat_k<<<G(NC), T, 0, stream>>>(bufA, adp, fseg, clusters, concat);
  gemm_bf16_k<<<GEMM_G, TG, 0, stream>>>(concat, wt_fuse, bufB, 256);
  colstats_k<<<COL_G, T, 0, stream>>>(bufB, fuse_gamma, fuse_beta, stats);
  bn_act_k<<<G(NC), T, 0, stream>>>(bufB, stats, nullptr, feat, fused, NC);  // +x_in after lrelu

  // ---- 2x 27-tap sparse conv (dominant compute) ----------------------------
  sconv_bf16_k<<<GEMM_G, TG, 0, stream>>>(fused, nidx, wt_c1, ctmp);
  colstats_k<<<COL_G, T, 0, stream>>>(ctmp, bn1_gamma, bn1_beta, stats);
  bn_act_k<<<G(NC), T, 0, stream>>>(ctmp, stats, nullptr, nullptr, d1, NC);
  sconv_bf16_k<<<GEMM_G, TG, 0, stream>>>(d1, nidx, wt_c2, ctmp);
  colstats_k<<<COL_G, T, 0, stream>>>(ctmp, bn2_gamma, bn2_beta, stats);
  bn_act_k<<<G(NC), T, 0, stream>>>(ctmp, stats, fused, nullptr, (float*)d_out, NC);  // +residual before lrelu
}
